// HSTUBlock_19267223289975
// MI455X (gfx1250) — compile-verified, matrix operand order checked by reading, not run on hardware
//
#include <hip/hip_runtime.h>
#include <hip/hip_bf16.h>
#include <math.h>

// Problem constants (match reference)
#define BQ     2
#define T_SEQ  2048
#define HDIM   1024
#define NHEAD  16
#define DHEAD  64
#define FFND   4096
#define MROWS  (BQ * T_SEQ)          // 4096

typedef __attribute__((ext_vector_type(16))) __bf16 v16bf;
typedef __attribute__((ext_vector_type(8)))  float  v8f;
typedef unsigned int v4u __attribute__((ext_vector_type(4)));
typedef int          v4i __attribute__((ext_vector_type(4)));
typedef int          v8i __attribute__((ext_vector_type(8)));

union BF16x16 { uint4 u[2]; v16bf v; };

// ---- Tensor Data Mover availability / arity handling ----------------------
#if defined(__has_builtin)
#  if __has_builtin(__builtin_amdgcn_tensor_load_to_lds) && \
      __has_builtin(__builtin_amdgcn_s_wait_tensorcnt)
#    define USE_TDM 1
#  endif
#endif
#ifndef USE_TDM
#  define USE_TDM 0
#endif

#if USE_TDM
#  if __has_include(<hip/amd_detail/amd_gfx1250_TDM.h>)
// amdgpu-toolchain (clang-23 / therock): 6-arg form
#    define TDM_CALL(g0, g1, g2, g3) \
       __builtin_amdgcn_tensor_load_to_lds((g0), (g1), (g2), (g3), \
           (v8i){0, 0, 0, 0, 0, 0, 0, 0}, 0)
#  else
// ROCm 7.2 (clang-22): 5-arg form
#    define TDM_CALL(g0, g1, g2, g3) \
       __builtin_amdgcn_tensor_load_to_lds((g0), (g1), (g2), (g3), 0)
#  endif

// Issue one 2D TDM tile load: tile [tile1 rows x tile0 elems] of a bf16
// tensor [nRows x kElems] (row-major, stride kElems), into LDS at ldsOff.
__device__ __forceinline__ void tdm_issue_2d(const unsigned short* gbase,
                                             long long elemOff, unsigned ldsOff,
                                             unsigned kElems, unsigned nRows,
                                             unsigned tile0, unsigned tile1) {
  unsigned long long ga =
      (unsigned long long)gbase + (unsigned long long)elemOff * 2ull;
  v4u g0 = {1u,                         // count=1, is_restore=0, gather=0
            ldsOff,                     // lds_addr
            (unsigned)ga,               // global_addr[31:0]
            (unsigned)((ga >> 32) & 0x01FFFFFFull) | (2u << 30)}; // addr hi | type=2
  unsigned long long s0 = kElems;       // tensor_dim0_stride (elements)
  v8i g1 = {
      (int)(1u << 16),                                      // data_size=1 (2B)
      (int)((kElems & 0xFFFFu) << 16),                      // dim0[15:0] @ [63:48]
      (int)((kElems >> 16) | ((nRows & 0xFFFFu) << 16)),    // dim0 hi | dim1 lo
      (int)((nRows >> 16) | (tile0 << 16)),                 // dim1 hi | tile_dim0
      (int)tile1,                                           // tile_dim1 (tile_dim2=0)
      (int)(s0 & 0xFFFFFFFFull),                            // stride0[31:0]
      (int)((s0 >> 32) & 0xFFFFull),                        // stride0[47:32]
      0};
  v4i z4 = {0, 0, 0, 0};
  TDM_CALL(g0, g1, z4, z4);
}
#endif

__device__ __forceinline__ unsigned short f2bf(float f) {
  union { float f; unsigned u; } v; v.f = f;
  unsigned r = 0x7FFFu + ((v.u >> 16) & 1u);
  return (unsigned short)((v.u + r) >> 16);
}
__device__ __forceinline__ float bf2f(unsigned short h) {
  union { unsigned u; float f; } v; v.u = ((unsigned)h) << 16;
  return v.f;
}
__device__ __forceinline__ float sig_(float x) { return 1.0f / (1.0f + __expf(-x)); }

// ---------------------------------------------------------------------------
// fp32 -> bf16 convert (weights)
// ---------------------------------------------------------------------------
__global__ __launch_bounds__(256) void cvt_bf16(const float* __restrict__ s,
                                                unsigned short* __restrict__ d, int n) {
  int i = blockIdx.x * 256 + threadIdx.x;
  if (i < n) d[i] = f2bf(s[i]);
}

// ---------------------------------------------------------------------------
// RMSNorm: z = rms_w * x * rsqrt(mean(x^2)+eps), bf16 out.  One block per row.
// ---------------------------------------------------------------------------
__global__ __launch_bounds__(256) void rmsnorm_k(const float* __restrict__ x,
                                                 const float* __restrict__ w,
                                                 unsigned short* __restrict__ z) {
  int row = blockIdx.x;
  const float* xr = x + (size_t)row * HDIM;
  int lane = threadIdx.x & 31, wave = threadIdx.x >> 5;
  float ss = 0.f;
  for (int i = threadIdx.x; i < HDIM; i += 256) { float v = xr[i]; ss += v * v; }
#pragma unroll
  for (int off = 16; off >= 1; off >>= 1) ss += __shfl_xor(ss, off, 32);
  __shared__ float red[8];
  if (lane == 0) red[wave] = ss;
  __syncthreads();
  float tot = 0.f;
#pragma unroll
  for (int i = 0; i < 8; ++i) tot += red[i];
  float inv = rsqrtf(tot / (float)HDIM + 1e-8f);
  for (int i = threadIdx.x; i < HDIM; i += 256)
    z[(size_t)row * HDIM + i] = f2bf(xr[i] * w[i] * inv);
}

// ---------------------------------------------------------------------------
// Generic WMMA bf16 GEMM:  C[M,N] = A[M,K] * W[N,K]^T   (both K-major)
// 256 threads = 8 waves (2x4), BM=BN=128, BK=32; each wave: 4x2 WMMA tiles.
// TDM double-buffered LDS staging (wave 0 issues, TENSORcnt-tracked).
// ---------------------------------------------------------------------------
#define BM 128
#define BN 128
#define BK 32

#define EPI_ROW_F32    0
#define EPI_ROW_BF16   1
#define EPI_HEAD_BF16  2   // out[b,h,t,d] bf16   (Q,K pre-RoPE)
#define EPI_VT_BF16    3   // out[b,h,d,t] bf16   (V transposed)
#define EPI_WO_RESID   4   // o = sig(res_w)*x + sig(gate_w)*y -> f32 + bf16
#define EPI_GELU_BF16  5   // bf16(gelu(v + bias[col]))
#define EPI_FFN2_OUT   6   // f32(resid + v + bias[col])

template <int EPI>
__global__ __launch_bounds__(256) void gemm_bf16(
    const unsigned short* __restrict__ A,   // [M,K] bf16
    const unsigned short* __restrict__ W,   // [N,K] bf16
    int M, int N, int K,
    float* __restrict__ outF, unsigned short* __restrict__ outB,
    const float* __restrict__ bias,
    const float* __restrict__ resid,
    const float* __restrict__ g_gate, const float* __restrict__ g_res) {
  __shared__ __align__(16) unsigned short lA[2][BM * BK];
  __shared__ __align__(16) unsigned short lB[2][BN * BK];
  const int tid  = threadIdx.x;
  const int lane = tid & 31;
  const int wave = tid >> 5;
  const int wm   = wave >> 2;         // 0..1
  const int wn   = wave & 3;          // 0..3
  const int half = lane >> 4;
  const int l15  = lane & 15;
  const int blockM = blockIdx.y * BM;
  const int blockN = blockIdx.x * BN;

  v8f acc[4][2];
#pragma unroll
  for (int mt = 0; mt < 4; ++mt)
#pragma unroll
    for (int nt = 0; nt < 2; ++nt)
      acc[mt][nt] = (v8f){0.f, 0.f, 0.f, 0.f, 0.f, 0.f, 0.f, 0.f};

  const int nk = K / BK;

#if USE_TDM
  const bool issuer = (wave == 0);
  if (issuer) {
    tdm_issue_2d(A, (long long)blockM * K, (unsigned)(size_t)&lA[0][0],
                 (unsigned)K, (unsigned)M, BK, BM);
    tdm_issue_2d(W, (long long)blockN * K, (unsigned)(size_t)&lB[0][0],
                 (unsigned)K, (unsigned)N, BK, BN);
  }
#endif

  for (int ki = 0; ki < nk; ++ki) {
    const int cur = ki & 1;
#if USE_TDM
    if (issuer) __builtin_amdgcn_s_wait_tensorcnt(0);
    __syncthreads();
    if (issuer && (ki + 1) < nk) {
      long long k0 = (long long)(ki + 1) * BK;
      tdm_issue_2d(A, (long long)blockM * K + k0,
                   (unsigned)(size_t)&lA[1 - cur][0], (unsigned)K, (unsigned)M, BK, BM);
      tdm_issue_2d(W, (long long)blockN * K + k0,
                   (unsigned)(size_t)&lB[1 - cur][0], (unsigned)K, (unsigned)N, BK, BN);
    }
#else
    __syncthreads();
    {
      int k0 = ki * BK;
#pragma unroll
      for (int c = tid; c < 512; c += 256) {
        int row = c >> 2, seg = c & 3;
        *(uint4*)&lA[cur][row * BK + seg * 8] =
            *(const uint4*)&A[(size_t)(blockM + row) * K + k0 + seg * 8];
        *(uint4*)&lB[cur][row * BK + seg * 8] =
            *(const uint4*)&W[(size_t)(blockN + row) * K + k0 + seg * 8];
      }
    }
    __syncthreads();
#endif

    const unsigned short* curA = &lA[cur][0];
    const unsigned short* curB = &lB[cur][0];
    // A fragments: lane<16 holds K{0..7,16..23}, lane>=16 holds K{8..15,24..31}
    BF16x16 af[4], bf[2];
#pragma unroll
    for (int mt = 0; mt < 4; ++mt) {
      const unsigned short* p = &curA[(wm * 64 + mt * 16 + l15) * BK];
      af[mt].u[0] = *(const uint4*)&p[half ? 8 : 0];
      af[mt].u[1] = *(const uint4*)&p[half ? 24 : 16];
    }
#pragma unroll
    for (int nt = 0; nt < 2; ++nt) {
      const unsigned short* p = &curB[(wn * 32 + nt * 16 + l15) * BK];
      bf[nt].u[0] = *(const uint4*)&p[half ? 8 : 0];
      bf[nt].u[1] = *(const uint4*)&p[half ? 24 : 16];
    }
#pragma unroll
    for (int mt = 0; mt < 4; ++mt)
#pragma unroll
      for (int nt = 0; nt < 2; ++nt)
        acc[mt][nt] = __builtin_amdgcn_wmma_f32_16x16x32_bf16(
            false, af[mt].v, false, bf[nt].v, (short)0, acc[mt][nt], false, false);
  }

  // Epilogue (C layout: lanes0-15 N=l15 rows 0-7 / lanes16-31 rows 8-15)
#pragma unroll
  for (int mt = 0; mt < 4; ++mt)
#pragma unroll
    for (int nt = 0; nt < 2; ++nt)
#pragma unroll
      for (int r = 0; r < 8; ++r) {
        int row = blockM + wm * 64 + mt * 16 + half * 8 + r;
        int col = blockN + wn * 32 + nt * 16 + l15;
        float v = acc[mt][nt][r];
        if (EPI == EPI_ROW_F32) {
          outF[(size_t)row * N + col] = v;
        } else if (EPI == EPI_ROW_BF16) {
          outB[(size_t)row * N + col] = f2bf(v);
        } else if (EPI == EPI_HEAD_BF16) {
          int b = row / T_SEQ, t = row % T_SEQ, h = col >> 6, d = col & 63;
          outB[(((size_t)b * NHEAD + h) * T_SEQ + t) * DHEAD + d] = f2bf(v);
        } else if (EPI == EPI_VT_BF16) {
          int b = row / T_SEQ, t = row % T_SEQ, h = col >> 6, d = col & 63;
          outB[(((size_t)b * NHEAD + h) * DHEAD + d) * T_SEQ + t] = f2bf(v);
        } else if (EPI == EPI_WO_RESID) {
          float o = sig_(g_res[0]) * resid[(size_t)row * HDIM + col] + sig_(g_gate[0]) * v;
          outF[(size_t)row * HDIM + col] = o;
          outB[(size_t)row * HDIM + col] = f2bf(o);
        } else if (EPI == EPI_GELU_BF16) {
          float t = v + bias[col];
          outB[(size_t)row * N + col] = f2bf(0.5f * t * (1.0f + erff(t * 0.70710678f)));
        } else if (EPI == EPI_FFN2_OUT) {
          outF[(size_t)row * HDIM + col] =
              resid[(size_t)row * HDIM + col] + v + bias[col];
        }
      }
}

// ---------------------------------------------------------------------------
// In-place RoPE on bf16 Q and K in [b,h,t,d] layout (positions = arange(T)).
// ---------------------------------------------------------------------------
__global__ __launch_bounds__(256) void rope_qk(unsigned short* __restrict__ Q,
                                               unsigned short* __restrict__ K) {
  size_t idx = (size_t)blockIdx.x * 256 + threadIdx.x;   // over B*NH*T*(D/2)
  int i = idx & 31;
  size_t row = idx >> 5;                                  // (b*NH+h)*T + t
  int t = (int)(row % T_SEQ);
  float f = (float)t * powf(10000.0f, -((float)(2 * i)) / 64.0f);
  float s, c;
  __sincosf(f, &s, &c);
  size_t base = row * DHEAD + 2 * i;
  float q0 = bf2f(Q[base]), q1 = bf2f(Q[base + 1]);
  Q[base]     = f2bf(q0 * c - q1 * s);
  Q[base + 1] = f2bf(q1 * c + q0 * s);
  float k0 = bf2f(K[base]), k1 = bf2f(K[base + 1]);
  K[base]     = f2bf(k0 * c - k1 * s);
  K[base + 1] = f2bf(k1 * c + k0 * s);
}

// ---------------------------------------------------------------------------
// Flash attention with rel-pos bias. Block = 4 waves; each wave owns 16 q rows.
// Q[b,h,t,d] bf16, K[b,h,t,d] bf16, Vt[b,h,d,t] bf16 -> pooled[b,t,h*64+d] f32
// ---------------------------------------------------------------------------
__global__ __launch_bounds__(128) void attn_k(const unsigned short* __restrict__ Qb,
                                              const unsigned short* __restrict__ Kb,
                                              const unsigned short* __restrict__ Vt,
                                              const float* __restrict__ rpt,
                                              float* __restrict__ pooled) {
  const int h = blockIdx.y, b = blockIdx.z;
  const int bh = b * NHEAD + h;
  const int wave = threadIdx.x >> 5;
  const int lane = threadIdx.x & 31;
  const int half = lane >> 4, l15 = lane & 15;
  const int qBase = blockIdx.x * 64 + wave * 16;

  __shared__ __align__(16) unsigned short Plds[4][16][64];

  // Q fragments (A layout), fixed for whole loop
  BF16x16 qf[2];
#pragma unroll
  for (int s = 0; s < 2; ++s) {
    const unsigned short* p = Qb + ((size_t)bh * T_SEQ + qBase + l15) * DHEAD + s * 32;
    qf[s].u[0] = *(const uint4*)&p[half ? 8 : 0];
    qf[s].u[1] = *(const uint4*)&p[half ? 24 : 16];
  }

  v8f o[4];
#pragma unroll
  for (int dt = 0; dt < 4; ++dt) o[dt] = (v8f){0.f,0.f,0.f,0.f,0.f,0.f,0.f,0.f};
  float m[8], l[8];
#pragma unroll
  for (int r = 0; r < 8; ++r) { m[r] = -1e30f; l[r] = 0.f; }

  for (int kt = 0; kt < T_SEQ; kt += 64) {
    // S = Q K^T  (4 column tiles of 16)
    v8f s4[4];
#pragma unroll
    for (int ct = 0; ct < 4; ++ct) {
      s4[ct] = (v8f){0.f,0.f,0.f,0.f,0.f,0.f,0.f,0.f};
      int kn = kt + ct * 16 + l15;
#pragma unroll
      for (int s = 0; s < 2; ++s) {
        BF16x16 kf;
        const unsigned short* p = Kb + ((size_t)bh * T_SEQ + kn) * DHEAD + s * 32;
        kf.u[0] = *(const uint4*)&p[half ? 8 : 0];
        kf.u[1] = *(const uint4*)&p[half ? 24 : 16];
        s4[ct] = __builtin_amdgcn_wmma_f32_16x16x32_bf16(
            false, qf[s].v, false, kf.v, (short)0, s4[ct], false, false);
      }
    }
    // scale + rel-pos bias + clip, row max
    float rowmax[8];
#pragma unroll
    for (int r = 0; r < 8; ++r) rowmax[r] = -1e30f;
#pragma unroll
    for (int ct = 0; ct < 4; ++ct)
#pragma unroll
      for (int r = 0; r < 8; ++r) {
        int qrow = qBase + half * 8 + r;
        int kcol = kt + ct * 16 + l15;
        int rel = kcol - qrow;
        rel = (rel < -16 ? -16 : (rel > 15 ? 15 : rel)) + 16;
        float sv = s4[ct][r] * 0.125f + rpt[rel];
        sv = fminf(fmaxf(sv, -50.f), 50.f);
        s4[ct][r] = sv;
        rowmax[r] = fmaxf(rowmax[r], sv);
      }
#pragma unroll
    for (int r = 0; r < 8; ++r)
#pragma unroll
      for (int off = 1; off <= 8; off <<= 1)
        rowmax[r] = fmaxf(rowmax[r], __shfl_xor(rowmax[r], off, 32));
    // online softmax update
    float alpha[8], rs[8];
#pragma unroll
    for (int r = 0; r < 8; ++r) {
      float mn = fmaxf(m[r], rowmax[r]);
      alpha[r] = __expf(m[r] - mn);
      m[r] = mn;
      rs[r] = 0.f;
    }
#pragma unroll
    for (int ct = 0; ct < 4; ++ct)
#pragma unroll
      for (int r = 0; r < 8; ++r) {
        float p = __expf(s4[ct][r] - m[r]);
        s4[ct][r] = p;
        rs[r] += p;
      }
#pragma unroll
    for (int r = 0; r < 8; ++r) {
#pragma unroll
      for (int off = 1; off <= 8; off <<= 1) rs[r] += __shfl_xor(rs[r], off, 32);
      l[r] = l[r] * alpha[r] + rs[r];
    }
#pragma unroll
    for (int dt = 0; dt < 4; ++dt)
#pragma unroll
      for (int r = 0; r < 8; ++r) o[dt][r] *= alpha[r];
    // transpose P through LDS into A layout
#pragma unroll
    for (int ct = 0; ct < 4; ++ct)
#pragma unroll
      for (int r = 0; r < 8; ++r)
        Plds[wave][half * 8 + r][ct * 16 + l15] = f2bf(s4[ct][r]);
    __syncthreads();
    BF16x16 pf[2];
#pragma unroll
    for (int s = 0; s < 2; ++s) {
      const unsigned short* p = &Plds[wave][l15][s * 32];
      pf[s].u[0] = *(const uint4*)&p[half ? 8 : 0];
      pf[s].u[1] = *(const uint4*)&p[half ? 24 : 16];
    }
    // O += P * V   (V as B operand from transposed layout: K-contig over t)
#pragma unroll
    for (int dt = 0; dt < 4; ++dt) {
      int d = dt * 16 + l15;
#pragma unroll
      for (int s = 0; s < 2; ++s) {
        BF16x16 vf;
        const unsigned short* p = Vt + ((size_t)bh * DHEAD + d) * T_SEQ + kt + s * 32;
        vf.u[0] = *(const uint4*)&p[half ? 8 : 0];
        vf.u[1] = *(const uint4*)&p[half ? 24 : 16];
        o[dt] = __builtin_amdgcn_wmma_f32_16x16x32_bf16(
            false, pf[s].v, false, vf.v, (short)0, o[dt], false, false);
      }
    }
    __syncthreads();
  }
  // normalize + store pooled row-layout
#pragma unroll
  for (int dt = 0; dt < 4; ++dt)
#pragma unroll
    for (int r = 0; r < 8; ++r) {
      int t = qBase + half * 8 + r;
      int d = dt * 16 + l15;
      pooled[((size_t)b * T_SEQ + t) * HDIM + h * DHEAD + d] = o[dt][r] / l[r];
    }
}

// ---------------------------------------------------------------------------
// gated = sigmoid(U) * pooled   (bf16 out)
// ---------------------------------------------------------------------------
__global__ __launch_bounds__(256) void gate_k(const unsigned short* __restrict__ U,
                                              const float* __restrict__ pooled,
                                              unsigned short* __restrict__ gated) {
  size_t i = (size_t)blockIdx.x * 256 + threadIdx.x;
  gated[i] = f2bf(sig_(bf2f(U[i])) * pooled[i]);
}

// ---------------------------------------------------------------------------
extern "C" void kernel_launch(void* const* d_in, const int* in_sizes, int n_in,
                              void* d_out, int out_size, void* d_ws, size_t ws_size,
                              hipStream_t stream) {
  const float* x      = (const float*)d_in[0];
  const float* wq     = (const float*)d_in[3];
  const float* wk     = (const float*)d_in[4];
  const float* wv     = (const float*)d_in[5];
  const float* wu     = (const float*)d_in[6];
  const float* wo     = (const float*)d_in[7];
  const float* f1w    = (const float*)d_in[8];
  const float* f1b    = (const float*)d_in[9];
  const float* f2w    = (const float*)d_in[10];
  const float* f2b    = (const float*)d_in[11];
  const float* rmsw   = (const float*)d_in[12];
  const float* rpt    = (const float*)d_in[13];
  const float* gate_w = (const float*)d_in[14];
  const float* res_w  = (const float*)d_in[15];

  char* ws = (char*)d_ws;
  size_t off = 0;
  auto carve = [&](size_t bytes) -> char* {
    off = (off + 255) & ~(size_t)255;
    char* p = ws + off;
    off += bytes;
    return p;
  };
  unsigned short* Z    = (unsigned short*)carve((size_t)MROWS * HDIM * 2);
  unsigned short* Wq   = (unsigned short*)carve((size_t)HDIM * HDIM * 2);
  unsigned short* Wk   = (unsigned short*)carve((size_t)HDIM * HDIM * 2);
  unsigned short* Wv   = (unsigned short*)carve((size_t)HDIM * HDIM * 2);
  unsigned short* Wu   = (unsigned short*)carve((size_t)HDIM * HDIM * 2);
  unsigned short* Wo   = (unsigned short*)carve((size_t)HDIM * HDIM * 2);
  unsigned short* W1   = (unsigned short*)carve((size_t)FFND * HDIM * 2);
  unsigned short* W2   = (unsigned short*)carve((size_t)HDIM * FFND * 2);
  unsigned short* Qb   = (unsigned short*)carve((size_t)MROWS * HDIM * 2);
  unsigned short* Kb   = (unsigned short*)carve((size_t)MROWS * HDIM * 2);
  unsigned short* Vt   = (unsigned short*)carve((size_t)MROWS * HDIM * 2);
  unsigned short* Ub   = (unsigned short*)carve((size_t)MROWS * HDIM * 2);
  float*          Pool = (float*)carve((size_t)MROWS * HDIM * 4);
  unsigned short* Gt   = (unsigned short*)carve((size_t)MROWS * HDIM * 2);
  float*          O1f  = (float*)carve((size_t)MROWS * HDIM * 4);
  unsigned short* O1b  = (unsigned short*)carve((size_t)MROWS * HDIM * 2);
  unsigned short* H1   = (unsigned short*)carve((size_t)MROWS * FFND * 2);

  // Weight converts
  cvt_bf16<<<(HDIM * HDIM + 255) / 256, 256, 0, stream>>>(wq, Wq, HDIM * HDIM);
  cvt_bf16<<<(HDIM * HDIM + 255) / 256, 256, 0, stream>>>(wk, Wk, HDIM * HDIM);
  cvt_bf16<<<(HDIM * HDIM + 255) / 256, 256, 0, stream>>>(wv, Wv, HDIM * HDIM);
  cvt_bf16<<<(HDIM * HDIM + 255) / 256, 256, 0, stream>>>(wu, Wu, HDIM * HDIM);
  cvt_bf16<<<(HDIM * HDIM + 255) / 256, 256, 0, stream>>>(wo, Wo, HDIM * HDIM);
  cvt_bf16<<<(FFND * HDIM + 255) / 256, 256, 0, stream>>>(f1w, W1, FFND * HDIM);
  cvt_bf16<<<(FFND * HDIM + 255) / 256, 256, 0, stream>>>(f2w, W2, FFND * HDIM);

  // RMSNorm
  rmsnorm_k<<<MROWS, 256, 0, stream>>>(x, rmsw, Z);

  // Projections
  dim3 gP(HDIM / BN, MROWS / BM);
  gemm_bf16<EPI_HEAD_BF16><<<gP, 256, 0, stream>>>(Z, Wq, MROWS, HDIM, HDIM,
      nullptr, Qb, nullptr, nullptr, nullptr, nullptr);
  gemm_bf16<EPI_HEAD_BF16><<<gP, 256, 0, stream>>>(Z, Wk, MROWS, HDIM, HDIM,
      nullptr, Kb, nullptr, nullptr, nullptr, nullptr);
  gemm_bf16<EPI_VT_BF16><<<gP, 256, 0, stream>>>(Z, Wv, MROWS, HDIM, HDIM,
      nullptr, Vt, nullptr, nullptr, nullptr, nullptr);
  gemm_bf16<EPI_ROW_BF16><<<gP, 256, 0, stream>>>(Z, Wu, MROWS, HDIM, HDIM,
      nullptr, Ub, nullptr, nullptr, nullptr, nullptr);

  // RoPE (in place, bf16)
  rope_qk<<<(BQ * NHEAD * T_SEQ * (DHEAD / 2)) / 256, 256, 0, stream>>>(Qb, Kb);

  // Flash attention
  attn_k<<<dim3(T_SEQ / 64, NHEAD, BQ), 128, 0, stream>>>(Qb, Kb, Vt, rpt, Pool);

  // Gating
  gate_k<<<(MROWS * HDIM) / 256, 256, 0, stream>>>(Ub, Pool, Gt);

  // Output projection + residual mix
  gemm_bf16<EPI_WO_RESID><<<gP, 256, 0, stream>>>(Gt, Wo, MROWS, HDIM, HDIM,
      O1f, O1b, nullptr, x, gate_w, res_w);

  // FFN1 (GeLU)
  dim3 gF1(FFND / BN, MROWS / BM);
  gemm_bf16<EPI_GELU_BF16><<<gF1, 256, 0, stream>>>(O1b, W1, MROWS, FFND, HDIM,
      nullptr, H1, f1b, nullptr, nullptr, nullptr);

  // FFN2 + bias + residual -> d_out
  gemm_bf16<EPI_FFN2_OUT><<<gP, 256, 0, stream>>>(H1, W2, MROWS, HDIM, FFND,
      (float*)d_out, nullptr, f2b, O1f, nullptr, nullptr);
}
